// StyleBlock_34540126995025
// MI455X (gfx1250) — matrix operand order repaired and checked by reading, not compiled, not run on hardware
//
#include <hip/hip_runtime.h>
#include <hip/hip_bf16.h>
#include <math.h>

typedef __attribute__((ext_vector_type(16))) __bf16 v16bf;
typedef __attribute__((ext_vector_type(8)))  float  v8f;

// Problem sizes (fixed by setup_inputs)
constexpr int Bn   = 32;    // batch
constexpr int Cin  = 128;   // in channels
constexpr int Cout = 128;   // out channels
constexpr int Zd   = 512;   // style z dim
constexpr int Hs   = 128;   // height
constexpr int Wd   = 128;   // width
constexpr int Kdim = Cin * 9;   // im2col K, tap-major: k = t*Cin + ci
constexpr int HW   = Hs * Wd;

// ---------------------------------------------------------------------------
// Kernel A: style = z @ W_style^T + b_style   ([32,512]x[128,512]^T -> [32,128])
// ---------------------------------------------------------------------------
__global__ void style_kernel(const float* __restrict__ z,
                             const float* __restrict__ Wsty,
                             const float* __restrict__ bsty,
                             float* __restrict__ style) {
  int gid = blockIdx.x * blockDim.x + threadIdx.x;   // 4096 = 32*128
  int b  = gid >> 7;
  int ci = gid & 127;
  const float4* zr = reinterpret_cast<const float4*>(z + (size_t)b * Zd);
  const float4* wr = reinterpret_cast<const float4*>(Wsty + (size_t)ci * Zd);
  float acc = bsty[ci];
#pragma unroll 4
  for (int j = 0; j < Zd / 4; ++j) {
    float4 a = zr[j], w = wr[j];
    acc += a.x * w.x + a.y * w.y + a.z * w.z + a.w * w.w;
  }
  style[gid] = acc;
}

// ---------------------------------------------------------------------------
// Kernel B: per-(b,co) modulate + demodulate, emit bf16 weights, TAP-MAJOR:
//   wmod[b][co][k] with k = (kh*3+kw)*128 + ci
// ---------------------------------------------------------------------------
__global__ void modw_kernel(const float* __restrict__ weight,  // [Cout][Cin][3][3]
                            const float* __restrict__ style,   // [B][Cin]
                            __bf16* __restrict__ wmod) {       // [B][Cout][Kdim]
  const int co  = blockIdx.x;
  const int b   = blockIdx.y;
  const int tid = threadIdx.x;        // 128 threads

  __shared__ float vbuf[Kdim];
  __shared__ float red[128];

  float p = 0.f;
  for (int idx = tid; idx < Kdim; idx += 128) {
    int ci = idx / 9;
    int t  = idx % 9;
    float v = weight[((size_t)co * Cin + ci) * 9 + t] * (style[b * Cin + ci] + 1.0f);
    vbuf[t * Cin + ci] = v;           // reorder to tap-major
    p += v * v;
  }
  red[tid] = p;
  __syncthreads();
  for (int s = 64; s > 0; s >>= 1) {
    if (tid < s) red[tid] += red[tid + s];
    __syncthreads();
  }
  float d = rsqrtf(red[0] + 1e-8f);

  __bf16* dst = wmod + ((size_t)(b * Cout + co)) * Kdim;
  for (int idx = tid; idx < Kdim; idx += 128)
    dst[idx] = (__bf16)(vbuf[idx] * d);
}

// ---------------------------------------------------------------------------
// Mish: x * tanh(softplus(x)); sp >= 0 so tanh via exp(-2sp) is stable.
// ---------------------------------------------------------------------------
__device__ __forceinline__ float mishf(float v) {
  float sp = (v > 20.f) ? v : log1pf(__expf(v));
  float e  = __expf(-2.f * sp);
  return v * ((1.f - e) / (1.f + e));
}

// ---------------------------------------------------------------------------
// Kernel C: implicit-GEMM grouped conv via bf16 WMMA, f32 accumulate.
// Block tile: M=128 (all c_out) x N=64 pixels (half of one image row).
// 8 waves; wave w owns rows [16w,16w+16), all 4 N-tiles of 16 pixels.
// ---------------------------------------------------------------------------
__global__ __launch_bounds__(256)
void conv_wmma_kernel(const float* __restrict__ x,      // [B][Cin][H][W] f32
                      const __bf16* __restrict__ wmod,  // [B][Cout][Kdim] bf16
                      float* __restrict__ out) {        // [B][Cout][H][W] f32
  constexpr int NT   = 64;    // pixel tile
  constexpr int CPAD = 136;   // ci padded (stride 272B = 68 dwords -> no bank conflicts)

  // x tile: rows h-1..h+1, cols w0-1..w0+64, ci innermost (bf16)
  __shared__ __bf16 xs[3][66][CPAD];

  const int half = blockIdx.x;
  const int h    = blockIdx.y;
  const int b    = blockIdx.z;
  const int w0   = half * NT;

  const int tid   = threadIdx.x;
  const int lane  = tid & 31;
  const int wave  = tid >> 5;
  const int lm    = lane & 15;   // row/col within 16
  const int lhalf = lane >> 4;   // 0 or 1

  // ---- stage x into LDS (fp32 -> bf16, zero-padded halo), ci innermost ----
  for (int idx = tid; idx < Cin * 3 * 66; idx += 256) {
    int c  = idx % 66;
    int r  = (idx / 66) % 3;
    int ci = idx / (66 * 3);
    int gr = h - 1 + r;
    int gc = w0 - 1 + c;
    float v = 0.f;
    if ((unsigned)gr < (unsigned)Hs && (unsigned)gc < (unsigned)Wd)
      v = x[(((size_t)b * Cin + ci) * Hs + gr) * Wd + gc];
    xs[r][c][ci] = (__bf16)v;
  }
  __syncthreads();

  const int m0 = wave * 16;
  v8f acc[4] = {v8f{}, v8f{}, v8f{}, v8f{}};

  // A fragment base: lane holds row m0+lm; K chunks {0..7,16..23} (lane<16)
  // or {8..15,24..31} (lane>=16), per the CDNA5 16-bit A layout.
  const __bf16* wrow = wmod + ((size_t)(b * Cout + m0 + lm)) * Kdim + lhalf * 8;

#pragma unroll
  for (int t = 0; t < 9; ++t) {
    const int dh = t / 3;
    const int dw = t % 3;
#pragma unroll
    for (int ci0 = 0; ci0 < Cin; ci0 += 32) {
      const int k0 = t * Cin + ci0;

      union { v16bf v; uint4 q[2]; } A;
      const uint4* wp = reinterpret_cast<const uint4*>(wrow + k0);
      A.q[0] = wp[0];   // K = k0 + (lhalf?8:0) .. +7
      A.q[1] = wp[2];   // K = k0 + 16 + (lhalf?8:0) .. +7

#pragma unroll
      for (int n = 0; n < 4; ++n) {
        // B fragment: lane holds column N = n*16+lm, ci chunks
        // {0..7,16..23} (lanes 0-15) or {8..15,24..31} (lanes 16-31), matching A.
        union { v16bf v; uint4 q[2]; } Bf;
        const uint4* xp = reinterpret_cast<const uint4*>(
            &xs[dh][n * 16 + lm + dw][ci0 + lhalf * 8]);
        Bf.q[0] = xp[0];
        Bf.q[1] = xp[2];
        acc[n] = __builtin_amdgcn_wmma_f32_16x16x32_bf16(
            false, A.v, false, Bf.v, (short)0, acc[n], false, false);
      }
    }
  }

  // ---- epilogue: mish + store (D VGPR r -> rows m0+r (lanes<16) / m0+r+8) ----
  float* orow = out + (size_t)(b * Cout) * HW + (size_t)h * Wd + w0;
#pragma unroll
  for (int n = 0; n < 4; ++n) {
    int pl = n * 16 + lm;
#pragma unroll
    for (int r = 0; r < 8; ++r) {
      int co = m0 + r + lhalf * 8;
      orow[(size_t)co * HW + pl] = mishf(acc[n][r]);
    }
  }
}

// ---------------------------------------------------------------------------
extern "C" void kernel_launch(void* const* d_in, const int* in_sizes, int n_in,
                              void* d_out, int out_size, void* d_ws, size_t ws_size,
                              hipStream_t stream) {
  const float* x      = (const float*)d_in[0];  // [32,128,128,128]
  const float* z      = (const float*)d_in[1];  // [32,512]
  const float* weight = (const float*)d_in[2];  // [128,128,3,3]
  const float* Wsty   = (const float*)d_in[3];  // [128,512]
  const float* bsty   = (const float*)d_in[4];  // [128]
  float* out = (float*)d_out;

  // workspace: style [32*128 f32] then wmod bf16 [32*128*1152]
  float*  style = (float*)d_ws;
  __bf16* wmod  = (__bf16*)((char*)d_ws + (size_t)Bn * Cin * sizeof(float));

  style_kernel<<<dim3((Bn * Cin) / 256), 256, 0, stream>>>(z, Wsty, bsty, style);
  modw_kernel<<<dim3(Cout, Bn), 128, 0, stream>>>(weight, style, wmod);
  conv_wmma_kernel<<<dim3(2, Hs, Bn), 256, 0, stream>>>(x, wmod, out);
}
